// EncoderBlock_13365938225270
// MI455X (gfx1250) — matrix operand order repaired and checked by reading, not compiled, not run on hardware
//
#include <hip/hip_runtime.h>
#include <hip/hip_bf16.h>
#include <stdint.h>

#define D_MODEL 1024
#define NHEAD   16
#define DKDIM   64
#define D_FF    4096
#define SEQ     2048
#define BATCH   2
#define MROWS   (BATCH*SEQ)   // 4096

typedef __attribute__((ext_vector_type(16))) __bf16 v16bf;
typedef __attribute__((ext_vector_type(8)))  float  v8f;
typedef __attribute__((ext_vector_type(4)))  int    v4i;

union FragU { uint4 u[2]; v16bf v; };

// CDNA5 async global->LDS path (builtin confirmed present; param0 = AS(1) v4i*,
// per round-2 diagnostic). Fallback = reg-staged pipeline if builtins absent.
#if defined(__AMDGCN__) && __has_builtin(__builtin_amdgcn_global_load_async_to_lds_b128) && \
    __has_builtin(__builtin_amdgcn_s_wait_asynccnt)
#define USE_ASYNC_LDS 1
#define ASYNC_CP_B128(gp, lp)                                                   \
  __builtin_amdgcn_global_load_async_to_lds_b128(                               \
      (__attribute__((address_space(1))) v4i*)(gp),                             \
      (__attribute__((address_space(3))) v4i*)(lp), 0, 0)
#else
#define USE_ASYNC_LDS 0
#endif

// round-to-nearest-even f32 -> bf16 (bit-level, no scalar __bf16 arithmetic)
__device__ __forceinline__ unsigned short f2bf(float f) {
  union { float f; unsigned u; } v; v.f = f;
  unsigned r = v.u + 0x7FFFu + ((v.u >> 16) & 1u);
  return (unsigned short)(r >> 16);
}

// ---------------------------------------------------------------- converts
__global__ void cvt_f32_bf16(const float* __restrict__ in,
                             unsigned short* __restrict__ out, int n) {
  int i = blockIdx.x * 256 + threadIdx.x;
  if (i < n) out[i] = f2bf(in[i]);
}

// ---------------------------------------------------------------- layernorm
// torch semantics: mean, unbiased std (ddof=1), eps added to std
__global__ __launch_bounds__(256)
void layernorm_bf16(const float* __restrict__ x,
                    const float* __restrict__ alpha,
                    const float* __restrict__ beta,
                    unsigned short* __restrict__ y) {
  __shared__ float s1[256], s2[256];
  int row = blockIdx.x;
  const float* xr = x + (size_t)row * D_MODEL;
  float vals[D_MODEL / 256];
  float a = 0.f, b = 0.f;
#pragma unroll
  for (int i = 0; i < D_MODEL / 256; ++i) {
    float v = xr[threadIdx.x + i * 256];
    vals[i] = v; a += v; b += v * v;
  }
  s1[threadIdx.x] = a; s2[threadIdx.x] = b;
  __syncthreads();
  for (int off = 128; off > 0; off >>= 1) {
    if (threadIdx.x < off) {
      s1[threadIdx.x] += s1[threadIdx.x + off];
      s2[threadIdx.x] += s2[threadIdx.x + off];
    }
    __syncthreads();
  }
  float mean = s1[0] * (1.f / D_MODEL);
  float var  = fmaxf((s2[0] - (float)D_MODEL * mean * mean) * (1.f / (D_MODEL - 1)), 0.f);
  float sc   = alpha[0] / (sqrtf(var) + 1e-6f);
  float be   = beta[0];
  unsigned short* yr = y + (size_t)row * D_MODEL;
#pragma unroll
  for (int i = 0; i < D_MODEL / 256; ++i)
    yr[threadIdx.x + i * 256] = f2bf((vals[i] - mean) * sc + be);
}

// ---------------------------------------------------------------- WMMA GEMM
// C[M,N] = A[M,K] (bf16 row-major) * Bw[N,K]^T (bf16 row-major: weight [out,in])
// block tile 128x128, 8 waves, wave tile 32x64, K step 32, double-buffered LDS.
enum { EPI_QK = 0, EPI_VT = 1, EPI_RESID_F32 = 2, EPI_RELU_BF16 = 3, EPI_BIAS_RESID_F32 = 4 };

#define LDSP 40   // padded half-stride: 80 B/row -> all fragment chunks 16B-aligned

__global__ __launch_bounds__(256)
void gemm_wmma_bf16(const unsigned short* __restrict__ A,
                    const unsigned short* __restrict__ Bw,
                    const float* __restrict__ bias,
                    const float* __restrict__ resid,
                    void* __restrict__ out,
                    int M, int N, int K, int mode) {
  __shared__ unsigned short As[2][128 * LDSP];
  __shared__ unsigned short Bs[2][128 * LDSP];
  const int tid  = threadIdx.x;
  const int lane = tid & 31, wid = tid >> 5;
  const int l15 = lane & 15, lh = lane >> 4;
  const int wm = (wid >> 1) * 32;       // 4 waves along M
  const int wn = (wid & 1) * 64;        // 2 waves along N
  const int bm = blockIdx.y * 128, bn = blockIdx.x * 128;

  // staging coords: each thread moves 2 A-chunks + 2 B-chunks (16B each) per K step
  const int r0 = tid >> 2, c0 = tid & 3;      // rows r0 and r0+64, halves c0*8..+8
  const unsigned short* ag0 = A  + (size_t)(bm + r0)      * K + c0 * 8;
  const unsigned short* ag1 = A  + (size_t)(bm + r0 + 64) * K + c0 * 8;
  const unsigned short* bg0 = Bw + (size_t)(bn + r0)      * K + c0 * 8;
  const unsigned short* bg1 = Bw + (size_t)(bn + r0 + 64) * K + c0 * 8;
  const int lo0 = r0 * LDSP + c0 * 8;
  const int lo1 = (r0 + 64) * LDSP + c0 * 8;

  v8f zero = {0.f,0.f,0.f,0.f,0.f,0.f,0.f,0.f};
  v8f acc[2][4];
#pragma unroll
  for (int i = 0; i < 2; ++i)
#pragma unroll
    for (int j = 0; j < 4; ++j) acc[i][j] = zero;

#if !USE_ASYNC_LDS
  uint4 ra0, ra1, rb0, rb1;
#endif
  int buf = 0;

#if USE_ASYNC_LDS
  // prologue: async-stage tile 0 (ASYNCcnt-tracked, no VGPR round-trip)
  ASYNC_CP_B128(ag0, &As[0][lo0]);
  ASYNC_CP_B128(ag1, &As[0][lo1]);
  ASYNC_CP_B128(bg0, &Bs[0][lo0]);
  ASYNC_CP_B128(bg1, &Bs[0][lo1]);
#else
  ra0 = *(const uint4*)ag0; ra1 = *(const uint4*)ag1;
  rb0 = *(const uint4*)bg0; rb1 = *(const uint4*)bg1;
#endif

  for (int kt = 0; kt < K; kt += 32) {
#if USE_ASYNC_LDS
    __builtin_amdgcn_s_wait_asynccnt(0);
    __syncthreads();
    if (kt + 32 < K) {   // stage next tile into the other buffer while computing
      ASYNC_CP_B128(ag0 + kt + 32, &As[buf ^ 1][lo0]);
      ASYNC_CP_B128(ag1 + kt + 32, &As[buf ^ 1][lo1]);
      ASYNC_CP_B128(bg0 + kt + 32, &Bs[buf ^ 1][lo0]);
      ASYNC_CP_B128(bg1 + kt + 32, &Bs[buf ^ 1][lo1]);
    }
#else
    // store previously-loaded tile, then immediately issue next global loads
    *(uint4*)&As[buf][lo0] = ra0;
    *(uint4*)&As[buf][lo1] = ra1;
    *(uint4*)&Bs[buf][lo0] = rb0;
    *(uint4*)&Bs[buf][lo1] = rb1;
    __syncthreads();
    if (kt + 32 < K) {
      ra0 = *(const uint4*)(ag0 + kt + 32);
      ra1 = *(const uint4*)(ag1 + kt + 32);
      rb0 = *(const uint4*)(bg0 + kt + 32);
      rb1 = *(const uint4*)(bg1 + kt + 32);
    }
    if (kt + 64 < K)
      __builtin_prefetch(ag0 + kt + 64, 0, 1);   // global_prefetch_b8
#endif

    // A fragment: lane = row, half-wave selects K {0-7,16-23} vs {8-15,24-31}
    FragU a[2];
#pragma unroll
    for (int mi = 0; mi < 2; ++mi) {
      const unsigned short* p = &As[buf][(wm + mi * 16 + l15) * LDSP + lh * 8];
      a[mi].u[0] = *(const uint4*)p;
      a[mi].u[1] = *(const uint4*)(p + 16);
    }
    // B fragment: lane = col, half-wave selects K {0-7,16-23} vs {8-15,24-31}
#pragma unroll
    for (int ni = 0; ni < 4; ++ni) {
      FragU b;
      const unsigned short* p = &Bs[buf][(wn + ni * 16 + l15) * LDSP + lh * 8];
      b.u[0] = *(const uint4*)p;
      b.u[1] = *(const uint4*)(p + 16);
#pragma unroll
      for (int mi = 0; mi < 2; ++mi)
        acc[mi][ni] = __builtin_amdgcn_wmma_f32_16x16x32_bf16(
            false, a[mi].v, false, b.v, (short)0, acc[mi][ni], false, false);
    }
    buf ^= 1;
  }

  // epilogue: C layout row = r + 8*half, col = lane&15
#pragma unroll
  for (int mi = 0; mi < 2; ++mi)
#pragma unroll
    for (int ni = 0; ni < 4; ++ni) {
      int col = bn + wn + ni * 16 + l15;
      float bv = (mode == EPI_RELU_BF16 || mode == EPI_BIAS_RESID_F32) ? bias[col] : 0.f;
#pragma unroll
      for (int r = 0; r < 8; ++r) {
        int row = bm + wm + mi * 16 + r + lh * 8;
        float v = acc[mi][ni][r];
        if (mode == EPI_QK) {            // -> [B,H,S,dk] bf16
          int bI = row >> 11, s = row & (SEQ - 1);
          int hh = col >> 6, d = col & 63;
          ((unsigned short*)out)[(((size_t)(bI * NHEAD + hh) * SEQ + s) << 6) + d] = f2bf(v);
        } else if (mode == EPI_VT) {     // -> [B,H,dk,S] bf16 (transposed V)
          int bI = row >> 11, s = row & (SEQ - 1);
          int hh = col >> 6, d = col & 63;
          ((unsigned short*)out)[((size_t)(bI * NHEAD + hh) * DKDIM + d) * SEQ + s] = f2bf(v);
        } else if (mode == EPI_RESID_F32) {
          size_t o = (size_t)row * N + col;
          ((float*)out)[o] = v + resid[o];
        } else if (mode == EPI_RELU_BF16) {
          float t = v + bv;
          ((unsigned short*)out)[(size_t)row * N + col] = f2bf(t > 0.f ? t : 0.f);
        } else {                          // EPI_BIAS_RESID_F32
          size_t o = (size_t)row * N + col;
          ((float*)out)[o] = v + bv + resid[o];
        }
      }
    }
}

// ---------------------------------------------------------------- flash attention
// one wave per 16-query strip; 32 keys per iteration; online softmax
__global__ __launch_bounds__(256)
void attention_wmma(const unsigned short* __restrict__ Q,   // [BH,S,dk]
                    const unsigned short* __restrict__ Kk,  // [BH,S,dk]
                    const unsigned short* __restrict__ Vt,  // [BH,dk,S]
                    const int* __restrict__ mask,           // [B,S]
                    unsigned short* __restrict__ Out) {     // [B,S,D] bf16
  __shared__ unsigned short psc[8][16 * LDSP];
  int lane = threadIdx.x & 31, wid = threadIdx.x >> 5;
  int l15 = lane & 15, lh = lane >> 4;
  int bh = blockIdx.x;            // 0..B*H-1
  int b = bh >> 4;
  int h = bh & 15;
  int q0 = blockIdx.y * 128 + wid * 16;

  const unsigned short* qb = Q  + (size_t)bh * SEQ * DKDIM;
  const unsigned short* kb = Kk + (size_t)bh * SEQ * DKDIM;
  const unsigned short* vb = Vt + (size_t)bh * DKDIM * SEQ;

  FragU qf[2];
#pragma unroll
  for (int dc = 0; dc < 2; ++dc) {
    const unsigned short* p = qb + (size_t)(q0 + l15) * DKDIM + dc * 32 + lh * 8;
    qf[dc].u[0] = *(const uint4*)p;
    qf[dc].u[1] = *(const uint4*)(p + 16);
  }

  v8f zero = {0.f,0.f,0.f,0.f,0.f,0.f,0.f,0.f};
  v8f o[4]; float mrow[8], lrow[8];
#pragma unroll
  for (int t = 0; t < 4; ++t) o[t] = zero;
#pragma unroll
  for (int r = 0; r < 8; ++r) { mrow[r] = -3.0e38f; lrow[r] = 0.f; }

  for (int ks = 0; ks < SEQ; ks += 32) {
    // scores 16x32 = Q(16x64) . K^T, two 16-col chunks, two 32-d chunks
    v8f s[2] = { zero, zero };
#pragma unroll
    for (int nc = 0; nc < 2; ++nc)
#pragma unroll
      for (int dc = 0; dc < 2; ++dc) {
        FragU kf;
        const unsigned short* p =
            kb + (size_t)(ks + nc * 16 + l15) * DKDIM + dc * 32 + lh * 8;
        kf.u[0] = *(const uint4*)p;
        kf.u[1] = *(const uint4*)(p + 16);
        s[nc] = __builtin_amdgcn_wmma_f32_16x16x32_bf16(
            false, qf[dc].v, false, kf.v, (short)0, s[nc], false, false);
      }

    // scale 1/sqrt(dk) + mask bias (per-lane column)
    float mb0 = (mask[b * SEQ + ks + l15]      == 0) ? -1e9f : 0.f;
    float mb1 = (mask[b * SEQ + ks + 16 + l15] == 0) ? -1e9f : 0.f;
#pragma unroll
    for (int r = 0; r < 8; ++r) {
      s[0][r] = s[0][r] * 0.125f + mb0;
      s[1][r] = s[1][r] * 0.125f + mb1;
    }

    // online softmax: row reductions across the 16 column-lanes of each half
    float corr[8], p0[8], p1[8];
#pragma unroll
    for (int r = 0; r < 8; ++r) {
      float t = fmaxf(s[0][r], s[1][r]);
      for (int off = 1; off < 16; off <<= 1) t = fmaxf(t, __shfl_xor(t, off, 32));
      float mnew = fmaxf(mrow[r], t);
      corr[r] = __expf(mrow[r] - mnew);
      p0[r] = __expf(s[0][r] - mnew);
      p1[r] = __expf(s[1][r] - mnew);
      float u = p0[r] + p1[r];
      for (int off = 1; off < 16; off <<= 1) u += __shfl_xor(u, off, 32);
      lrow[r] = lrow[r] * corr[r] + u;
      mrow[r] = mnew;
    }
#pragma unroll
    for (int t = 0; t < 4; ++t)
#pragma unroll
      for (int r = 0; r < 8; ++r) o[t][r] *= corr[r];

    // repack P (C layout) -> A-fragment layout via per-wave LDS scratch
    unsigned short* sc = psc[wid];
#pragma unroll
    for (int r = 0; r < 8; ++r) {
      int rr = r + lh * 8;
      sc[rr * LDSP + l15]      = f2bf(p0[r]);
      sc[rr * LDSP + 16 + l15] = f2bf(p1[r]);
    }
    FragU pf;
    {
      const unsigned short* p = &sc[l15 * LDSP + lh * 8];
      pf.u[0] = *(const uint4*)p;
      pf.u[1] = *(const uint4*)(p + 16);
    }

    // O(16x64) += P(16x32) . V(32x64); V^T makes B-frag loads contiguous
#pragma unroll
    for (int nc = 0; nc < 4; ++nc) {
      FragU vf;
      const unsigned short* p =
          vb + (size_t)(nc * 16 + l15) * SEQ + ks + lh * 8;
      vf.u[0] = *(const uint4*)p;
      vf.u[1] = *(const uint4*)(p + 16);
      o[nc] = __builtin_amdgcn_wmma_f32_16x16x32_bf16(
          false, pf.v, false, vf.v, (short)0, o[nc], false, false);
    }
  }

  // normalize and store [B,S,D] bf16 for the Wo GEMM
#pragma unroll
  for (int r = 0; r < 8; ++r) lrow[r] = 1.f / lrow[r];
#pragma unroll
  for (int nc = 0; nc < 4; ++nc) {
    int d = h * 64 + nc * 16 + l15;
#pragma unroll
    for (int r = 0; r < 8; ++r) {
      int qrow = q0 + r + lh * 8;
      Out[(size_t)(b * SEQ + qrow) * D_MODEL + d] = f2bf(o[nc][r] * lrow[r]);
    }
  }
}

// ---------------------------------------------------------------- launcher
extern "C" void kernel_launch(void* const* d_in, const int* in_sizes, int n_in,
                              void* d_out, int out_size, void* d_ws, size_t ws_size,
                              hipStream_t stream) {
  const float* src  = (const float*)d_in[0];
  const int*   mask = (const int*)  d_in[1];
  const float* wq   = (const float*)d_in[2];
  const float* wk   = (const float*)d_in[3];
  const float* wv   = (const float*)d_in[4];
  const float* wo   = (const float*)d_in[5];
  const float* w1   = (const float*)d_in[6];
  const float* b1   = (const float*)d_in[7];
  const float* w2   = (const float*)d_in[8];
  const float* b2   = (const float*)d_in[9];
  const float* a1   = (const float*)d_in[10];
  const float* be1  = (const float*)d_in[11];
  const float* a2   = (const float*)d_in[12];
  const float* be2  = (const float*)d_in[13];
  float* out = (float*)d_out;

  // carve workspace (256B aligned)
  char* w = (char*)d_ws;
  auto carve = [&](size_t bytes) { char* p = w; w += (bytes + 255) & ~(size_t)255; return p; };
  unsigned short* wqb  = (unsigned short*)carve((size_t)D_MODEL * D_MODEL * 2);
  unsigned short* wkb  = (unsigned short*)carve((size_t)D_MODEL * D_MODEL * 2);
  unsigned short* wvb  = (unsigned short*)carve((size_t)D_MODEL * D_MODEL * 2);
  unsigned short* wob  = (unsigned short*)carve((size_t)D_MODEL * D_MODEL * 2);
  unsigned short* w1b  = (unsigned short*)carve((size_t)D_FF * D_MODEL * 2);
  unsigned short* w2b  = (unsigned short*)carve((size_t)D_MODEL * D_FF * 2);
  unsigned short* xln1 = (unsigned short*)carve((size_t)MROWS * D_MODEL * 2);
  unsigned short* qbuf = (unsigned short*)carve((size_t)MROWS * D_MODEL * 2);
  unsigned short* kbuf = (unsigned short*)carve((size_t)MROWS * D_MODEL * 2);
  unsigned short* vtbf = (unsigned short*)carve((size_t)MROWS * D_MODEL * 2);
  unsigned short* attn = (unsigned short*)carve((size_t)MROWS * D_MODEL * 2);
  float*          src2 = (float*)         carve((size_t)MROWS * D_MODEL * 4);
  unsigned short* xln2 = (unsigned short*)carve((size_t)MROWS * D_MODEL * 2);
  unsigned short* hbuf = (unsigned short*)carve((size_t)MROWS * D_FF * 2);

  const int n1 = D_MODEL * D_MODEL, n2 = D_FF * D_MODEL;
  cvt_f32_bf16<<<(n1 + 255) / 256, 256, 0, stream>>>(wq, wqb, n1);
  cvt_f32_bf16<<<(n1 + 255) / 256, 256, 0, stream>>>(wk, wkb, n1);
  cvt_f32_bf16<<<(n1 + 255) / 256, 256, 0, stream>>>(wv, wvb, n1);
  cvt_f32_bf16<<<(n1 + 255) / 256, 256, 0, stream>>>(wo, wob, n1);
  cvt_f32_bf16<<<(n2 + 255) / 256, 256, 0, stream>>>(w1, w1b, n2);
  cvt_f32_bf16<<<(n2 + 255) / 256, 256, 0, stream>>>(w2, w2b, n2);

  layernorm_bf16<<<MROWS, 256, 0, stream>>>(src, a1, be1, xln1);

  dim3 gD(D_MODEL / 128, MROWS / 128);   // (8, 32)
  gemm_wmma_bf16<<<gD, 256, 0, stream>>>(xln1, wqb, nullptr, nullptr, qbuf,
                                         MROWS, D_MODEL, D_MODEL, EPI_QK);
  gemm_wmma_bf16<<<gD, 256, 0, stream>>>(xln1, wkb, nullptr, nullptr, kbuf,
                                         MROWS, D_MODEL, D_MODEL, EPI_QK);
  gemm_wmma_bf16<<<gD, 256, 0, stream>>>(xln1, wvb, nullptr, nullptr, vtbf,
                                         MROWS, D_MODEL, D_MODEL, EPI_VT);

  attention_wmma<<<dim3(BATCH * NHEAD, SEQ / 128), 256, 0, stream>>>(
      qbuf, kbuf, vtbf, mask, attn);

  gemm_wmma_bf16<<<gD, 256, 0, stream>>>(attn, wob, nullptr, src, src2,
                                         MROWS, D_MODEL, D_MODEL, EPI_RESID_F32);

  layernorm_bf16<<<MROWS, 256, 0, stream>>>(src2, a2, be2, xln2);

  dim3 gF(D_FF / 128, MROWS / 128);      // (32, 32)
  gemm_wmma_bf16<<<gF, 256, 0, stream>>>(xln2, w1b, b1, nullptr, hbuf,
                                         MROWS, D_FF, D_MODEL, EPI_RELU_BF16);
  gemm_wmma_bf16<<<gD, 256, 0, stream>>>(hbuf, w2b, b2, src2, out,
                                         MROWS, D_MODEL, D_FF, EPI_BIAS_RESID_F32);
  (void)in_sizes; (void)n_in; (void)out_size; (void)ws_size;
}
